// MultiHeadAttention_12558484373945
// MI455X (gfx1250) — matrix operand order repaired and checked by reading, not compiled, not run on hardware
//
#include <hip/hip_runtime.h>

// Problem constants (B,S,D,H) = (4, 2048, 1024, 16), DK=64, TEMP=sqrt(64)=8.
#define B_  4
#define S_  2048
#define D_  1024
#define H_  16
#define DK_ 64

typedef __attribute__((ext_vector_type(16))) _Float16 v16h;
typedef __attribute__((ext_vector_type(8)))  _Float16 v8h;
typedef __attribute__((ext_vector_type(8)))  float    v8f;
typedef __attribute__((ext_vector_type(4)))  float    v4f;

// A-fragment loader for 16-bit 16x32 A: per-lane 16 contiguous halfs at p
// (K klo..klo+15).
static __device__ inline v16h load_a16(const _Float16* p) {
  v8h lo = *(const v8h*)(p);
  v8h hi = *(const v8h*)(p + 8);
  v16h r;
#pragma unroll
  for (int i = 0; i < 8; ++i) { r[i] = lo[i]; r[i + 8] = hi[i]; }
  return r;
}

static __device__ inline v8f wmma_f16(v16h a, v16h b, v8f c) {
  return __builtin_amdgcn_wmma_f32_16x16x32_f16(false, a, false, b, (short)0, c,
                                                false, false);
}

// Async copy of one 16B chunk from global to LDS (per active lane).
// VDST = 32-bit LDS byte offset (low 32 bits of the generic pointer per the
// flat->LDS aperture rule), VADDR = 64-bit global address.  Tracked by
// ASYNCcnt; completion enforced with s_wait_asynccnt below.
static __device__ inline void async_g2l_b128(const _Float16* g, _Float16* l) {
  unsigned lofs = (unsigned)(size_t)l;
  asm volatile("global_load_async_to_lds_b128 %0, %1, off"
               :: "v"(lofs), "v"(g) : "memory");
}

static __device__ inline void wait_async_le4() {
#if __has_builtin(__builtin_amdgcn_s_wait_asynccnt)
  __builtin_amdgcn_s_wait_asynccnt(4);
#else
  asm volatile("s_wait_asynccnt 0x4" ::: "memory");
#endif
}
static __device__ inline void wait_async_le0() {
#if __has_builtin(__builtin_amdgcn_s_wait_asynccnt)
  __builtin_amdgcn_s_wait_asynccnt(0);
#else
  asm volatile("s_wait_asynccnt 0x0" ::: "memory");
#endif
}

// ---------------------------------------------------------------------------
// Stage 0: pack W (D x D f32, k-major) -> Wt (N x K f16), optionally scaled.
// ---------------------------------------------------------------------------
__global__ __launch_bounds__(256) void pack_wt_kernel(
    const float* __restrict__ W, _Float16* __restrict__ Wt, float scale) {
  const int idx = blockIdx.x * 256 + threadIdx.x;   // over D*D
  const int n = idx >> 10;
  const int k = idx & (D_ - 1);
  Wt[idx] = (_Float16)(W[k * D_ + n] * scale);
}

// ---------------------------------------------------------------------------
// Stage 1: projection GEMM  out = x @ W  (via Wt), one wave per 16x64 tile.
// vmode==0 : store f16 [ (b*H+h)*S + s ][dk]   (Q / K layout)
// vmode==1 : store f16 [ (b*H+h)*DK + dk ][s]  (V transposed layout)
// ---------------------------------------------------------------------------
__global__ __launch_bounds__(128) void proj_kernel(
    const float* __restrict__ x, const _Float16* __restrict__ Wt,
    _Float16* __restrict__ outp, int vmode) {
  __shared__ _Float16 lds[4][16 * 64];
  const int wave  = threadIdx.x >> 5;
  const int lane  = threadIdx.x & 31;
  const int lrow  = lane & 15;
  const int lhalf = lane >> 4;
  const int wgid  = blockIdx.x * 4 + wave;
  const int mt = wgid >> 4;          // 0..511 row tiles over B*S
  const int h  = wgid & 15;          // head == 64-col tile
  const int m0 = mt * 16;
  const int n0 = h * 64;
  const int b  = m0 / S_;
  const int s0 = m0 % S_;

  const float* arow = x + (size_t)(m0 + lrow) * D_;
  v8f acc[4] = {};
#pragma unroll 1
  for (int k0 = 0; k0 < D_; k0 += 32) {
    const int klo = k0 + lhalf * 16;
    v4f f0 = *(const v4f*)(arow + klo);
    v4f f1 = *(const v4f*)(arow + klo + 4);
    v4f f2 = *(const v4f*)(arow + klo + 8);
    v4f f3 = *(const v4f*)(arow + klo + 12);
    v16h a;
#pragma unroll
    for (int i = 0; i < 4; ++i) {
      a[i]      = (_Float16)f0[i];
      a[i + 4]  = (_Float16)f1[i];
      a[i + 8]  = (_Float16)f2[i];
      a[i + 12] = (_Float16)f3[i];
    }
#pragma unroll
    for (int nt = 0; nt < 4; ++nt) {
      const _Float16* brow =
          Wt + (size_t)(n0 + nt * 16 + lrow) * D_ + k0 + lhalf * 16;
      v16h bb = *(const v16h*)brow;               // B: K = 16*lhalf + e
      acc[nt] = wmma_f16(a, bb, acc[nt]);
    }
  }

  if (vmode) {
    // lane holds a fixed dk column, rows s0+8*lhalf+r contiguous -> b128 store
    const size_t vbase = ((size_t)(b * H_ + h) * DK_) * S_;
#pragma unroll
    for (int nt = 0; nt < 4; ++nt) {
      const int dk = nt * 16 + lrow;
      v8h hv;
#pragma unroll
      for (int r = 0; r < 8; ++r) hv[r] = (_Float16)acc[nt][r];
      *(v8h*)(outp + vbase + (size_t)dk * S_ + s0 + lhalf * 8) = hv;
    }
  } else {
    // stage through per-wave LDS, then coalesced 16B row-chunk stores
    _Float16* t = lds[wave];
#pragma unroll
    for (int nt = 0; nt < 4; ++nt)
#pragma unroll
      for (int r = 0; r < 8; ++r)
        t[(r + 8 * lhalf) * 64 + nt * 16 + lrow] = (_Float16)acc[nt][r];
    __builtin_amdgcn_wave_barrier();
    const size_t qbase = ((size_t)(b * H_ + h) * S_ + s0) * DK_;
#pragma unroll
    for (int p = 0; p < 4; ++p) {
      const int c = p * 32 + lane;
      const int row = c >> 3;
      const int off = (c & 7) * 8;
      *(v8h*)(outp + qbase + (size_t)row * DK_ + off) =
          *(const v8h*)(t + row * 64 + off);
    }
  }
}

// ---------------------------------------------------------------------------
// Stage 2: flash attention. One wave per (b,h, 16-query tile); 32 keys/iter.
// The 4 waves of a block share (b,h): K/V tiles are staged once per block
// into LDS with double-buffered global_load_async_to_lds_b128 (ASYNCcnt).
// TEMP already folded into Q (Wq scaled by 8). ctx stored f16 [b][s][h*64+dk].
// ---------------------------------------------------------------------------
__global__ __launch_bounds__(128) void attn_kernel(
    const _Float16* __restrict__ Qh, const _Float16* __restrict__ Kh,
    const _Float16* __restrict__ Vt, const int* __restrict__ mask,
    _Float16* __restrict__ ctx) {
  __shared__ _Float16 kbuf[2][32 * 64];   // [key-in-tile][dk]      4KB each
  __shared__ _Float16 vbuf[2][64 * 32];   // [dk][key-in-tile]      4KB each
  __shared__ _Float16 pstage[4][16 * 64]; // per-wave P / ctx staging
  const int tid   = threadIdx.x;
  const int wave  = tid >> 5;
  const int lane  = tid & 31;
  const int lrow  = lane & 15;
  const int lhalf = lane >> 4;
  const int wgid  = blockIdx.x * 4 + wave;
  const int qt = wgid & 127;
  const int bh = wgid >> 7;
  const int b  = bh >> 4;
  const int h  = bh & 15;
  const int q0 = qt * 16;

  const _Float16* Qb = Qh + (size_t)bh * S_ * DK_;
  const _Float16* Kb = Kh + (size_t)bh * S_ * DK_;
  const _Float16* Vb = Vt + (size_t)bh * DK_ * S_;
  const int*      mb = mask + (size_t)b * S_ * S_;

  const int klo = lhalf * 16;
  const _Float16* qrow = Qb + (size_t)(q0 + lrow) * DK_;
  v16h aq0 = load_a16(qrow + klo);        // Q[:, 0:32]
  v16h aq1 = load_a16(qrow + 32 + klo);   // Q[:, 32:64]

  float mrun[8], lrun[8];
#pragma unroll
  for (int r = 0; r < 8; ++r) { mrun[r] = -3.0e38f; lrun[r] = 0.0f; }
  v8f o[4] = {};
  _Float16* t = pstage[wave];

  // per-thread chunk assignments for the async tile copies (2 chunks each
  // for K and V; 256 x 16B chunks per 4KB tile, 128 threads)
  // prologue: stage tile kb=0 into buffer 0
#pragma unroll
  for (int j = 0; j < 2; ++j) {
    const int c = tid + j * 128;
    async_g2l_b128(Kb + (size_t)c * 8, &kbuf[0][c * 8]);
    const int vr = c >> 2, vo = (c & 3) * 8;
    async_g2l_b128(Vb + (size_t)vr * S_ + vo, &vbuf[0][vr * 32 + vo]);
  }

  for (int kb = 0; kb < S_; kb += 32) {
    const int cur = (kb >> 5) & 1;
    if (kb + 32 < S_) {
      const int nxt = cur ^ 1;
#pragma unroll
      for (int j = 0; j < 2; ++j) {
        const int c = tid + j * 128;
        async_g2l_b128(Kb + (size_t)(kb + 32) * DK_ + c * 8,
                       &kbuf[nxt][c * 8]);
        const int vr = c >> 2, vo = (c & 3) * 8;
        async_g2l_b128(Vb + (size_t)vr * S_ + (kb + 32) + vo,
                       &vbuf[nxt][vr * 32 + vo]);
      }
      wait_async_le4();   // 8 outstanding; oldest 4 (current tile) done
    } else {
      wait_async_le0();
    }
    __syncthreads();      // current tile visible to all waves

    // scores for two 16-key blocks: C = Q(16x64) x K^T(64x16), 2 WMMAs each
    v8f sc[2];
#pragma unroll
    for (int g = 0; g < 2; ++g) {
      const int kk = g * 16 + lrow;  // key-in-tile (B column)
      v16h bk0 = *(const v16h*)(&kbuf[cur][kk * 64 + lhalf * 16]);
      v16h bk1 = *(const v16h*)(&kbuf[cur][kk * 64 + 32 + lhalf * 16]);
      v8f c = {};
      c = wmma_f16(aq0, bk0, c);
      c = wmma_f16(aq1, bk1, c);
      sc[g] = c;
    }
    // reference semantics: scores = -1e9 where mask==0 (mask fits in L2)
#pragma unroll
    for (int g = 0; g < 2; ++g) {
      const int key = kb + g * 16 + lrow;
#pragma unroll
      for (int r = 0; r < 8; ++r) {
        const int qr = q0 + r + 8 * lhalf;
        if (mb[(size_t)qr * S_ + key] == 0) sc[g][r] = -1.0e9f;
      }
    }
    // online softmax: per-row reductions stay inside each 16-lane half
#pragma unroll
    for (int r = 0; r < 8; ++r) {
      float v = fmaxf(sc[0][r], sc[1][r]);
#pragma unroll
      for (int off = 1; off < 16; off <<= 1)
        v = fmaxf(v, __shfl_xor(v, off, 32));
      const float mnew  = fmaxf(mrun[r], v);
      const float alpha = __expf(mrun[r] - mnew);
      mrun[r] = mnew;
      const float p0 = __expf(sc[0][r] - mnew);
      const float p1 = __expf(sc[1][r] - mnew);
      sc[0][r] = p0; sc[1][r] = p1;
      float s = p0 + p1;
#pragma unroll
      for (int off = 1; off < 16; off <<= 1) s += __shfl_xor(s, off, 32);
      lrun[r] = lrun[r] * alpha + s;
#pragma unroll
      for (int nt = 0; nt < 4; ++nt) o[nt][r] *= alpha;
    }
    // C-layout -> A-layout via per-wave LDS (in-order DS within a wave)
#pragma unroll
    for (int r = 0; r < 8; ++r) {
      const int row = r + 8 * lhalf;
      t[row * 32 + lrow]      = (_Float16)sc[0][r];
      t[row * 32 + 16 + lrow] = (_Float16)sc[1][r];
    }
    __builtin_amdgcn_wave_barrier();
    v16h pa = load_a16(t + lrow * 32 + klo);
    __builtin_amdgcn_wave_barrier();
    // O += P(16x32) x V(32x64); vbuf rows are key-contiguous
#pragma unroll
    for (int nt = 0; nt < 4; ++nt) {
      const int dk = nt * 16 + lrow;  // B column
      v16h bv = *(const v16h*)(&vbuf[cur][dk * 32 + lhalf * 16]);
      o[nt] = wmma_f16(pa, bv, o[nt]);
    }
    __syncthreads();      // all waves done with buf[cur] before next overwrite
  }

  // normalize, stage to LDS, coalesced f16 stores into ctx[b][s][h*64+dk]
  float inv[8];
#pragma unroll
  for (int r = 0; r < 8; ++r) inv[r] = 1.0f / lrun[r];
#pragma unroll
  for (int nt = 0; nt < 4; ++nt)
#pragma unroll
    for (int r = 0; r < 8; ++r)
      t[(r + 8 * lhalf) * 64 + nt * 16 + lrow] =
          (_Float16)(o[nt][r] * inv[r]);
  __builtin_amdgcn_wave_barrier();
  const size_t cbase = (size_t)(b * S_ + q0) * D_ + (size_t)h * DK_;
#pragma unroll
  for (int p = 0; p < 4; ++p) {
    const int c = p * 32 + lane;
    const int row = c >> 3;
    const int off = (c & 7) * 8;
    *(v8h*)(ctx + cbase + (size_t)row * D_ + off) =
        *(const v8h*)(t + row * 64 + off);
  }
}

// ---------------------------------------------------------------------------
// Stage 3: out = ctx @ Wo + residual, then LayerNorm over D.
// 16 waves per block own the same 16 rows; wave w covers columns [64w,64w+64).
// ---------------------------------------------------------------------------
__global__ __launch_bounds__(512) void out_ln_kernel(
    const _Float16* __restrict__ ctx, const _Float16* __restrict__ Wot,
    const float* __restrict__ resid, const float* __restrict__ gamma,
    const float* __restrict__ beta, float* __restrict__ out) {
  __shared__ float ssum[16];
  __shared__ float ssq[16];
  const int wave  = threadIdx.x >> 5;
  const int lane  = threadIdx.x & 31;
  const int lrow  = lane & 15;
  const int lhalf = lane >> 4;
  const int m0 = blockIdx.x * 16;
  const int n0 = wave * 64;
  if (threadIdx.x < 16) { ssum[threadIdx.x] = 0.0f; ssq[threadIdx.x] = 0.0f; }
  __syncthreads();

  const _Float16* arow = ctx + (size_t)(m0 + lrow) * D_;
  v8f acc[4] = {};
#pragma unroll 1
  for (int k0 = 0; k0 < D_; k0 += 32) {
    v16h a = load_a16(arow + k0 + lhalf * 16);
#pragma unroll
    for (int nt = 0; nt < 4; ++nt) {
      const _Float16* brow =
          Wot + (size_t)(n0 + nt * 16 + lrow) * D_ + k0 + lhalf * 16;
      v16h bb = *(const v16h*)brow;
      acc[nt] = wmma_f16(a, bb, acc[nt]);
    }
  }

  // residual + per-row partial stats; ds_add_f32 combines the 16 waves
  float x[4][8];
#pragma unroll
  for (int r = 0; r < 8; ++r) {
    const size_t row = m0 + r + 8 * lhalf;
    float s = 0.0f, q2 = 0.0f;
#pragma unroll
    for (int nt = 0; nt < 4; ++nt) {
      const int n = n0 + nt * 16 + lrow;
      const float vv = acc[nt][r] + resid[row * D_ + n];
      x[nt][r] = vv;
      s += vv; q2 += vv * vv;
    }
#pragma unroll
    for (int off = 1; off < 16; off <<= 1) {
      s  += __shfl_xor(s, off, 32);
      q2 += __shfl_xor(q2, off, 32);
    }
    if (lrow == 0) {                       // lanes 0 and 16 (one per half)
      atomicAdd(&ssum[r + 8 * lhalf], s);
      atomicAdd(&ssq[r + 8 * lhalf], q2);
    }
  }
  __syncthreads();

#pragma unroll
  for (int r = 0; r < 8; ++r) {
    const int rl = r + 8 * lhalf;
    const float mu   = ssum[rl] * (1.0f / D_);
    const float var  = ssq[rl] * (1.0f / D_) - mu * mu;
    const float rstd = rsqrtf(var + 1e-6f);
    const size_t row = m0 + rl;
#pragma unroll
    for (int nt = 0; nt < 4; ++nt) {
      const int n = n0 + nt * 16 + lrow;
      out[row * D_ + n] = (x[nt][r] - mu) * rstd * gamma[n] + beta[n];
    }
  }
}

// ---------------------------------------------------------------------------
extern "C" void kernel_launch(void* const* d_in, const int* in_sizes, int n_in,
                              void* d_out, int out_size, void* d_ws,
                              size_t ws_size, hipStream_t stream) {
  (void)in_sizes; (void)n_in; (void)out_size; (void)ws_size;
  const float* q     = (const float*)d_in[0];
  const float* k     = (const float*)d_in[1];
  const float* v     = (const float*)d_in[2];
  const int*   mask  = (const int*)d_in[3];
  const float* Wq    = (const float*)d_in[4];
  const float* Wk    = (const float*)d_in[5];
  const float* Wv    = (const float*)d_in[6];
  const float* Wo    = (const float*)d_in[7];
  const float* gamma = (const float*)d_in[8];
  const float* beta  = (const float*)d_in[9];
  float* out = (float*)d_out;

  char* ws = (char*)d_ws;
  const size_t WT  = (size_t)D_ * D_ * sizeof(_Float16);       // 2 MB each
  const size_t ACT = (size_t)B_ * S_ * D_ * sizeof(_Float16);  // 16 MB each
  _Float16* Wqt  = (_Float16*)(ws + 0 * WT);
  _Float16* Wkt  = (_Float16*)(ws + 1 * WT);
  _Float16* Wvt  = (_Float16*)(ws + 2 * WT);
  _Float16* Wot  = (_Float16*)(ws + 3 * WT);
  _Float16* Qhp  = (_Float16*)(ws + 4 * WT);
  _Float16* Khp  = (_Float16*)(ws + 4 * WT + 1 * ACT);
  _Float16* Vtp  = (_Float16*)(ws + 4 * WT + 2 * ACT);
  _Float16* ctxp = (_Float16*)(ws + 4 * WT + 3 * ACT);         // total 72 MB

  const int packBlocks = (D_ * D_) / 256;
  pack_wt_kernel<<<packBlocks, 256, 0, stream>>>(Wq, Wqt, 8.0f);  // fold TEMP
  pack_wt_kernel<<<packBlocks, 256, 0, stream>>>(Wk, Wkt, 1.0f);
  pack_wt_kernel<<<packBlocks, 256, 0, stream>>>(Wv, Wvt, 1.0f);
  pack_wt_kernel<<<packBlocks, 256, 0, stream>>>(Wo, Wot, 1.0f);

  // (B*S/16) * (D/64) = 8192 waves, 4 waves/block
  proj_kernel<<<2048, 128, 0, stream>>>(q, Wqt, Qhp, 0);
  proj_kernel<<<2048, 128, 0, stream>>>(k, Wkt, Khp, 0);
  proj_kernel<<<2048, 128, 0, stream>>>(v, Wvt, Vtp, 1);

  // B*H * S/16 = 8192 waves; 4 waves/block share one (b,h)
  attn_kernel<<<2048, 128, 0, stream>>>(Qhp, Khp, Vtp, mask, ctxp);

  // B*S/16 = 512 blocks of 16 waves
  out_ln_kernel<<<512, 512, 0, stream>>>(ctxp, Wot, q, gamma, beta, out);
}
